// ATDGeo_35098472743146
// MI455X (gfx1250) — matrix operand order repaired and checked by reading, not compiled, not run on hardware
//
#include <hip/hip_runtime.h>
#include <hip/hip_bf16.h>

typedef __attribute__((ext_vector_type(16))) _Float16 v16h;
typedef __attribute__((ext_vector_type(8)))  float    v8f;

#define WS_     16
#define N_TOK   256
#define NHEAD   6
#define HD      32
#define DIM     192
#define C3      576      // 3*DIM
#define NB      256      // batch windows
#define QSCALE  0.17677669529663687f   // 32^-0.5

// ---------------------------------------------------------------------------
// Kernel 0: flatten rel-pos bias per head: rpbh[h*65536 + i*256 + j]
//           = table[rpi[i,j]*6 + h]   (removes dependent gather from hot loop)
// ---------------------------------------------------------------------------
__global__ __launch_bounds__(256)
void rpb_kernel(const int* __restrict__ rpi,
                const float* __restrict__ table,
                float* __restrict__ rpbh) {
    const int idx = blockIdx.x * 256 + threadIdx.x;     // 0..65535
    const int t = rpi[idx];
    #pragma unroll
    for (int h = 0; h < NHEAD; ++h)
        rpbh[(size_t)h * (N_TOK * N_TOK) + idx] = table[t * NHEAD + h];
}

// ---------------------------------------------------------------------------
// Kernel 1: per-batch geo cumsum tables ct_h[16][16], ct_v[16][16]
//   ct ws layout: ctws[b*512 + idx] = ct_h, ctws[b*512 + 256 + idx] = ct_v
// ---------------------------------------------------------------------------
__global__ __launch_bounds__(256)
void geo_ct_kernel(const float* __restrict__ feat,
                   const float* __restrict__ sigma_p,
                   float* __restrict__ ctws) {
    const int b = blockIdx.x;
    const int t = threadIdx.x;            // 0..255 = r*16+c
    const int r = t >> 4, c = t & 15;
    const float sig = fabsf(sigma_p[0]);
    const float* fb = feat + (size_t)b * WS_ * WS_ * DIM;

    __shared__ float ih[256], iv[256], chs[256], cvs[256];

    float dh = 0.f, dv = 0.f;
    const float* fc  = fb + (size_t)(r * 16 + c) * DIM;
    if (c > 0) {
        const float* fl = fc - DIM;                      // (r, c-1)
        #pragma unroll 4
        for (int ch = 0; ch < DIM; ++ch) dh += fabsf(fc[ch] - fl[ch]);
    }
    if (r > 0) {
        const float* fu = fc - 16 * DIM;                 // (r-1, c)
        #pragma unroll 4
        for (int ch = 0; ch < DIM; ++ch) dv += fabsf(fc[ch] - fu[ch]);
    }
    ih[t] = (c > 0) ? (1.0f + sig * dh) : 0.0f;
    iv[t] = (r > 0) ? (1.0f + sig * dv) : 0.0f;
    __syncthreads();

    if (t < 16) {                    // horizontal inclusive cumsum along c, row t
        float run = 0.f;
        for (int cc = 0; cc < 16; ++cc) { run += ih[t * 16 + cc]; chs[t * 16 + cc] = run; }
    } else if (t < 32) {             // vertical inclusive cumsum along r, col t-16
        const int cc = t - 16;
        float run = 0.f;
        for (int rr = 0; rr < 16; ++rr) { run += iv[rr * 16 + cc]; cvs[rr * 16 + cc] = run; }
    }
    __syncthreads();

    ctws[(size_t)b * 512 + t]       = chs[t];
    ctws[(size_t)b * 512 + 256 + t] = cvs[t];
}

// ---------------------------------------------------------------------------
// Kernel 2: fused attention per (b, h) with online softmax, WMMA f16->f32.
//   8 waves; wave w owns rows [32w, 32w+32) (2 M-tiles).
//   Writes x to xbuf[(b*256 + i)*192 + h*32 + d]  (f32)
// ---------------------------------------------------------------------------
__global__ __launch_bounds__(256)
void attn_kernel(const float* __restrict__ qkv,
                 const float* __restrict__ rpbh,
                 const float* __restrict__ maskg,
                 const float* __restrict__ geo_scale,
                 const float* __restrict__ ctws,
                 float*       __restrict__ xbuf) {
    const int b   = blockIdx.x;
    const int h   = blockIdx.y;
    const int tid = threadIdx.x;
    const int w   = tid >> 5;            // wave id 0..7
    const int lane = tid & 31;
    const int lg  = lane >> 4;           // lane group 0/1
    const int ln  = lane & 15;

    __shared__ _Float16 sK[N_TOK * HD];       // K row-major [token][d]     16 KB
    __shared__ _Float16 sVt[HD * N_TOK];      // V transposed [d][token]    16 KB
    __shared__ _Float16 sP[8 * 2 * 16 * 32];  // per-wave P staging         16 KB
    __shared__ float    sCth[256], sCtv[256];

    // ---- stage K, Vt (f32 -> f16), ct tables ----
    {
        const float* kg = qkv + (size_t)b * N_TOK * C3 + DIM + h * HD;
        const float* vg = kg + DIM;
        const int i = tid;   // one token per thread
        #pragma unroll
        for (int d = 0; d < HD; ++d) {
            sK [i * HD + d] = (_Float16)kg[(size_t)i * C3 + d];
            sVt[d * N_TOK + i] = (_Float16)vg[(size_t)i * C3 + d];
        }
        sCth[tid] = ctws[(size_t)b * 512 + tid];
        sCtv[tid] = ctws[(size_t)b * 512 + 256 + tid];
    }
    __syncthreads();

    // ---- hoisted Q A-fragments (2 M-tiles per wave) ----
    // A 16x32 f16, contiguous k order used here (B matches): lane m=ln holds K = lg*16 + e (e<16)
    const float* qg = qkv + (size_t)b * N_TOK * C3 + h * HD;
    v16h aq[2];
    #pragma unroll
    for (int mt = 0; mt < 2; ++mt) {
        const int i = (w * 2 + mt) * 16 + ln;
        const float* qr = qg + (size_t)i * C3;
        const int k0 = lg * 16;
        #pragma unroll
        for (int e = 0; e < 8; ++e) {
            aq[mt][e]     = (_Float16)(qr[k0 + e]      * QSCALE);
            aq[mt][8 + e] = (_Float16)(qr[k0 + 8 + e] * QSCALE);
        }
    }

    const float gs = geo_scale[h];
    const float* maskb  = maskg + (size_t)b * N_TOK * N_TOK;
    const float* rpbh_h = rpbh  + (size_t)h * N_TOK * N_TOK;

    float mrun[2][8], lrun[2][8];
    v8f   o[2][2];
    #pragma unroll
    for (int mt = 0; mt < 2; ++mt) {
        #pragma unroll
        for (int r = 0; r < 8; ++r) { mrun[mt][r] = -1e30f; lrun[mt][r] = 0.f; }
        o[mt][0] = (v8f){}; o[mt][1] = (v8f){};
    }

    _Float16* sPw = sP + w * 1024;    // 2 tiles * 16 * 32 halfs

    // warm the first column block's bias lines (one line per lane-row)
    {
        const int prow = w * 32 + lane;              // 32 rows of this wave
        __builtin_prefetch(&maskb [prow * N_TOK], 0, 1);
        __builtin_prefetch(&rpbh_h[prow * N_TOK], 0, 1);
    }

    for (int jb = 0; jb < 8; ++jb) {          // 32 columns per step
        const int jt0 = jb * 2;
        // prefetch next column block's mask/rpb rows (lane -> distinct row line)
        if (jb < 7) {
            const int prow = w * 32 + lane;
            __builtin_prefetch(&maskb [prow * N_TOK + (jb + 1) * 32], 0, 1);
            __builtin_prefetch(&rpbh_h[prow * N_TOK + (jb + 1) * 32], 0, 1);
        }
        // B fragments for K^T tiles: contiguous v16h in sK
        const v16h* bk0 = (const v16h*)(sK + ((jt0 * 16 + ln) * HD + lg * 16));
        const v16h* bk1 = (const v16h*)(sK + (((jt0 + 1) * 16 + ln) * HD + lg * 16));
        // B fragments for V: contiguous v16h in sVt
        const v16h* bv0 = (const v16h*)(sVt + (ln       * N_TOK + jb * 32 + lg * 16));
        const v16h* bv1 = (const v16h*)(sVt + ((16 + ln) * N_TOK + jb * 32 + lg * 16));

        #pragma unroll
        for (int mt = 0; mt < 2; ++mt) {
            v8f zacc = (v8f){};
            v8f s0 = __builtin_amdgcn_wmma_f32_16x16x32_f16(false, aq[mt], false, *bk0,
                                                            (short)0, zacc, false, false);
            v8f s1 = __builtin_amdgcn_wmma_f32_16x16x32_f16(false, aq[mt], false, *bk1,
                                                            (short)0, zacc, false, false);

            // ---- fused bias: rpb + mask - geo*gs (independent flat loads) ----
            const int ibase = (w * 2 + mt) * 16 + 8 * lg;
            float a0[8], a1[8], tmax[8], tsum[8], fr[8];
            #pragma unroll
            for (int r = 0; r < 8; ++r) {
                const int i  = ibase + r;
                const int ri = i >> 4, ci = i & 15;
                const float chi = sCth[ri * 16 + ci];
                const int j0 = jb * 32 + ln;
                const int j1 = j0 + 16;
                {
                    const int rj = j0 >> 4, cj = j0 & 15;
                    float geo = fabsf(sCth[ri * 16 + cj] - chi)
                              + fabsf(sCtv[rj * 16 + cj] - sCtv[ri * 16 + cj]);
                    float bias = rpbh_h[i * N_TOK + j0]
                               - geo * gs + maskb[i * N_TOK + j0];
                    a0[r] = s0[r] + bias;
                }
                {
                    const int rj = j1 >> 4, cj = j1 & 15;
                    float geo = fabsf(sCth[ri * 16 + cj] - chi)
                              + fabsf(sCtv[rj * 16 + cj] - sCtv[ri * 16 + cj]);
                    float bias = rpbh_h[i * N_TOK + j1]
                               - geo * gs + maskb[i * N_TOK + j1];
                    a1[r] = s1[r] + bias;
                }
                tmax[r] = fmaxf(a0[r], a1[r]);
            }
            // row-max across the 16 lanes of each half-wave
            #pragma unroll
            for (int r = 0; r < 8; ++r)
                #pragma unroll
                for (int off = 8; off >= 1; off >>= 1)
                    tmax[r] = fmaxf(tmax[r], __shfl_xor(tmax[r], off, 32));
            // online softmax update
            #pragma unroll
            for (int r = 0; r < 8; ++r) {
                const float nm = fmaxf(mrun[mt][r], tmax[r]);
                fr[r] = __expf(mrun[mt][r] - nm);
                a0[r] = __expf(a0[r] - nm);
                a1[r] = __expf(a1[r] - nm);
                tsum[r] = a0[r] + a1[r];
                mrun[mt][r] = nm;
            }
            #pragma unroll
            for (int r = 0; r < 8; ++r)
                #pragma unroll
                for (int off = 8; off >= 1; off >>= 1)
                    tsum[r] += __shfl_xor(tsum[r], off, 32);
            #pragma unroll
            for (int r = 0; r < 8; ++r) {
                lrun[mt][r] = lrun[mt][r] * fr[r] + tsum[r];
                o[mt][0][r] *= fr[r];
                o[mt][1][r] *= fr[r];
            }
            // ---- stage P tile (16x32) to wave-local LDS as f16 ----
            _Float16* sPmt = sPw + mt * 512;
            #pragma unroll
            for (int r = 0; r < 8; ++r) {
                const int row = r + 8 * lg;
                sPmt[row * 32 + ln]      = (_Float16)a0[r];
                sPmt[row * 32 + 16 + ln] = (_Float16)a1[r];
            }
            // re-read P in A-fragment layout (same wave; DS ops in-order)
            v16h ap = {};
            const int k0 = lg * 16;
            #pragma unroll
            for (int e = 0; e < 8; ++e) {
                ap[e]     = sPmt[ln * 32 + k0 + e];
                ap[8 + e] = sPmt[ln * 32 + k0 + 8 + e];
            }
            o[mt][0] = __builtin_amdgcn_wmma_f32_16x16x32_f16(false, ap, false, *bv0,
                                                              (short)0, o[mt][0], false, false);
            o[mt][1] = __builtin_amdgcn_wmma_f32_16x16x32_f16(false, ap, false, *bv1,
                                                              (short)0, o[mt][1], false, false);
        }
    }

    // ---- normalize and write x (b, n, h*32+d) ----
    #pragma unroll
    for (int mt = 0; mt < 2; ++mt) {
        #pragma unroll
        for (int dh = 0; dh < 2; ++dh) {
            #pragma unroll
            for (int r = 0; r < 8; ++r) {
                const int i = (w * 2 + mt) * 16 + r + 8 * lg;
                const int d = dh * 16 + ln;
                const float val = o[mt][dh][r] / lrun[mt][r];
                xbuf[((size_t)b * N_TOK + i) * DIM + h * HD + d] = val;
            }
        }
    }
}

// ---------------------------------------------------------------------------
// Kernel 3: projection GEMM  out[t, co] = sum_k x[t,k]*W[co,k] + b[co]
//   65536 x 192 @ (192x192)^T, WMMA f16 tiles, 128 tokens per block.
// ---------------------------------------------------------------------------
__global__ __launch_bounds__(256)
void proj_kernel(const float* __restrict__ xbuf,
                 const float* __restrict__ pw,
                 const float* __restrict__ pb,
                 float*       __restrict__ out) {
    const int tid  = threadIdx.x;
    const int w    = tid >> 5;
    const int lane = tid & 31;
    const int lg   = lane >> 4;
    const int ln   = lane & 15;
    const int tbase = blockIdx.x * 128 + w * 16;

    // hoist A fragments for all 6 K-tiles (K = 192 = 6*32)
    v16h ax[6];
    const float* xr = xbuf + (size_t)(tbase + ln) * DIM;
    #pragma unroll
    for (int kt = 0; kt < 6; ++kt) {
        const int k0 = kt * 32 + lg * 16;
        #pragma unroll
        for (int e = 0; e < 8; ++e) {
            ax[kt][e]     = (_Float16)xr[k0 + e];
            ax[kt][8 + e] = (_Float16)xr[k0 + 8 + e];
        }
    }

    for (int nt = 0; nt < 12; ++nt) {
        const int co = nt * 16 + ln;
        v8f acc = (v8f){};
        #pragma unroll
        for (int kt = 0; kt < 6; ++kt) {
            v16h bw = {};
            const float* wr = pw + (size_t)co * DIM + kt * 32 + lg * 16;
            #pragma unroll
            for (int e = 0; e < 16; ++e) bw[e] = (_Float16)wr[e];
            acc = __builtin_amdgcn_wmma_f32_16x16x32_f16(false, ax[kt], false, bw,
                                                         (short)0, acc, false, false);
        }
        const float bias = pb[co];
        #pragma unroll
        for (int r = 0; r < 8; ++r) {
            const int t = tbase + r + 8 * lg;
            out[(size_t)t * DIM + co] = acc[r] + bias;
        }
    }
}

// ---------------------------------------------------------------------------
extern "C" void kernel_launch(void* const* d_in, const int* in_sizes, int n_in,
                              void* d_out, int out_size, void* d_ws, size_t ws_size,
                              hipStream_t stream) {
    const float* qkv   = (const float*)d_in[0];
    const float* feat  = (const float*)d_in[1];
    const int*   rpi   = (const int*)  d_in[2];
    const float* maskg = (const float*)d_in[3];
    const float* table = (const float*)d_in[4];
    const float* gsc   = (const float*)d_in[5];
    const float* pw    = (const float*)d_in[6];
    const float* pb    = (const float*)d_in[7];
    const float* sigma = (const float*)d_in[8];
    float* out = (float*)d_out;

    float* ctws = (float*)d_ws;                       // 256*512 floats   = 512 KB
    float* rpbh = ctws + (size_t)NB * 512;            // 6*65536 floats   = 1.5 MB
    float* xbuf = rpbh + (size_t)NHEAD * N_TOK * N_TOK; // 12.58M floats  = 48 MB

    rpb_kernel<<<N_TOK, 256, 0, stream>>>(rpi, table, rpbh);
    geo_ct_kernel<<<NB, 256, 0, stream>>>(feat, sigma, ctws);
    attn_kernel<<<dim3(NB, NHEAD), 256, 0, stream>>>(qkv, rpbh, maskg, gsc, ctws, xbuf);
    proj_kernel<<<(NB * N_TOK) / 128, 256, 0, stream>>>(xbuf, pw, pb, out);
}
